// DeepSeekMoE_46359876993614
// MI455X (gfx1250) — matrix operand order repaired and checked by reading, not compiled, not run on hardware
//
#include <hip/hip_runtime.h>
#include <hip/hip_bf16.h>
#include <cstdint>
#include <cstddef>

#define TOK 4096   // B*S
#define HD  1024   // H
#define ID  2816   // I
#define NE  7      // routed experts
#define KT  64     // K elements staged per barrier (2 WMMA k-steps)

typedef __attribute__((ext_vector_type(16))) __bf16 v16bf;
typedef __attribute__((ext_vector_type(8)))  __bf16 v8bf;
typedef __attribute__((ext_vector_type(8)))  float  v8f;
typedef __attribute__((ext_vector_type(4)))  int    v4i;

// Pointee types in explicit address spaces for the async-LDS builtin
// (clang-22 signature: (AS1 v4i*, AS3 v4i*, imm offset, imm cpol)).
typedef __attribute__((address_space(1))) v4i as1_v4i;
typedef __attribute__((address_space(3))) v4i as3_v4i;

#if __has_builtin(__builtin_amdgcn_global_load_async_to_lds_b128) && \
    __has_builtin(__builtin_amdgcn_s_wait_asynccnt)
#define ASYNC_LDS 1
#else
#define ASYNC_LDS 0
#endif

// 16-byte global -> LDS copy; async (ASYNCcnt) when available on this toolchain.
static __device__ __forceinline__ void cp16(void* lds_dst, const void* gsrc) {
#if ASYNC_LDS
    __builtin_amdgcn_global_load_async_to_lds_b128(
        (as1_v4i*)(uintptr_t)gsrc,
        (as3_v4i*)(uintptr_t)lds_dst,
        0, 0);
#else
    *(uint4*)lds_dst = *(const uint4*)gsrc;
#endif
}

static __device__ __forceinline__ void cp_wait() {
#if ASYNC_LDS
    __builtin_amdgcn_s_wait_asynccnt(0);
#endif
}

static __device__ __forceinline__ float fast_rcp(float x) {
#if __has_builtin(__builtin_amdgcn_rcpf)
    return __builtin_amdgcn_rcpf(x);
#else
    return 1.f / x;
#endif
}

// ---------------------------------------------------------------- utilities

__global__ __launch_bounds__(32)
void k_init_counts(int* counts) {
    if (threadIdx.x < 8) counts[threadIdx.x] = 0;
}

__global__ __launch_bounds__(256)
void k_cvt_bf16(const float* __restrict__ x, __bf16* __restrict__ xb, int n) {
    int i = (blockIdx.x * 256 + threadIdx.x) * 4;
    if (i + 3 < n) {
        float4 v = *(const float4*)(x + i);
        xb[i + 0] = (__bf16)v.x;
        xb[i + 1] = (__bf16)v.y;
        xb[i + 2] = (__bf16)v.z;
        xb[i + 3] = (__bf16)v.w;
    }
}

// ------------------------------------------------------------------- router

__global__ __launch_bounds__(256)
void k_router(const float* __restrict__ x, const float* __restrict__ rw,
              const float* __restrict__ rbias, int* __restrict__ counts,
              int* __restrict__ lists, float* __restrict__ wts,
              int* __restrict__ slots) {
    int lane = threadIdx.x & 31;
    int t = blockIdx.x * 8 + (threadIdx.x >> 5);
    float acc[NE];
#pragma unroll
    for (int e = 0; e < NE; e++) acc[e] = 0.f;
    for (int hh = lane; hh < HD; hh += 32) {
        float xv = x[(size_t)t * HD + hh];
        const float* rp = rw + (size_t)hh * NE;
#pragma unroll
        for (int e = 0; e < NE; e++) acc[e] += xv * rp[e];
    }
#pragma unroll
    for (int e = 0; e < NE; e++) {
#pragma unroll
        for (int off = 16; off > 0; off >>= 1)
            acc[e] += __shfl_xor(acc[e], off, 32);
    }
    if (lane == 0) {
        float p[NE];
#pragma unroll
        for (int e = 0; e < NE; e++)
            p[e] = 1.f / (1.f + __expf(-(acc[e] + rbias[e])));
        int i0 = 0;
#pragma unroll
        for (int e = 1; e < NE; e++) if (p[e] > p[i0]) i0 = e;
        int i1 = (i0 == 0) ? 1 : 0;
#pragma unroll
        for (int e = 0; e < NE; e++)
            if (e != i0 && p[e] > p[i1]) i1 = e;
        float s  = p[i0] + p[i1];
        float w0 = p[i0] / s, w1 = p[i1] / s;
        int pos0 = atomicAdd(&counts[i0], 1);
        lists[i0 * TOK + pos0] = t; wts[i0 * TOK + pos0] = w0; slots[i0 * TOK + pos0] = 0;
        int pos1 = atomicAdd(&counts[i1], 1);
        lists[i1 * TOK + pos1] = t; wts[i1 * TOK + pos1] = w1; slots[i1 * TOK + pos1] = 1;
    }
}

// ------------------------------------------------------- fragment builder
// A fragment (16x32 bf16): lane L (m=L%16, kh=L/16) holds
// K = {kh*16 .. kh*16+15} (contiguous, matches B fragments)
static __device__ __forceinline__ v16bf make_afrag(const __bf16* p, int khalf) {
    v8bf lo = *(const v8bf*)(p + khalf * 16);
    v8bf hi = *(const v8bf*)(p + khalf * 16 + 8);
    v16bf r;
#pragma unroll
    for (int i = 0; i < 8; i++) { r[i] = lo[i]; r[8 + i] = hi[i]; }
    return r;
}

// ----------------------------------------------------- GEMM 1: gate+up+silu
// block tile M=128 (gathered tokens) x N=64 (of I); 8 waves (4 M x 2 N),
// each wave: 2x2 16x16 C tiles for BOTH gate and up. K staged 64 at a time.

__global__ __launch_bounds__(256)
void k_gemm_gateup(const __bf16* __restrict__ xb,
                   const float* __restrict__ gw, const float* __restrict__ uw,
                   const int* __restrict__ list, const int* __restrict__ cnt_ptr,
                   __bf16* __restrict__ hout) {
    int tid = threadIdx.x;
    int cnt = cnt_ptr ? *cnt_ptr : TOK;
    int row0 = blockIdx.y * 128;
    if (row0 >= cnt || cnt == 0) return;
    int n0 = blockIdx.x * 64;

    __shared__ __align__(32) __bf16 As[128 * KT];  // [m][k], stride KT
    __shared__ __align__(32) __bf16 Bg[64 * KT];   // transposed: [n][k]
    __shared__ __align__(32) __bf16 Bu[64 * KT];

    int lane = tid & 31;
    int wave = tid >> 5;
    int wm = wave & 3;
    int wn = wave >> 2;
    int mrow = lane & 15;
    int khalf = lane >> 4;

    v8f accG[2][2], accU[2][2];
#pragma unroll
    for (int a = 0; a < 2; a++)
#pragma unroll
        for (int b = 0; b < 2; b++) {
            v8f z = {0.f, 0.f, 0.f, 0.f, 0.f, 0.f, 0.f, 0.f};
            accG[a][b] = z; accU[a][b] = z;
        }

    // A loader: 2 threads per row, 32 bf16 (64B) each -> 4x b128
    int ar = tid >> 1;
    int ah = (tid & 1) * 32;
    int aridx = row0 + ar; if (aridx >= cnt) aridx = cnt - 1;
    int atok = list ? list[aridx] : aridx;
    // B loader: 4 threads per k-row, 16 floats each
    int bk = tid >> 2;
    int bq = (tid & 3) * 16;

    for (int k0 = 0; k0 < HD; k0 += KT) {
        const float* gp = gw + (size_t)(k0 + bk) * ID + n0 + bq;
        const float* up = uw + (size_t)(k0 + bk) * ID + n0 + bq;
        float4 g[4], u[4];
#pragma unroll
        for (int c = 0; c < 4; c++) {
            g[c] = ((const float4*)gp)[c];
            u[c] = ((const float4*)up)[c];
        }
        if (k0 + KT < HD) {
            __builtin_prefetch(gw + (size_t)(k0 + KT + bk) * ID + n0 + bq, 0, 1);
            __builtin_prefetch(uw + (size_t)(k0 + KT + bk) * ID + n0 + bq, 0, 1);
        }
        __syncthreads();   // previous stage's LDS reads complete
        // async A tile: gathered token rows -> LDS
        {
            const __bf16* asrc = xb + (size_t)atok * HD + k0 + ah;
            __bf16* adst = As + ar * KT + ah;
#pragma unroll
            for (int c = 0; c < 4; c++)
                cp16(adst + c * 8, asrc + c * 8);
        }
        // B tiles: fp32 -> bf16, transposed [n][k]
#pragma unroll
        for (int c = 0; c < 4; c++) {
            float gv[4] = {g[c].x, g[c].y, g[c].z, g[c].w};
            float uv[4] = {u[c].x, u[c].y, u[c].z, u[c].w};
#pragma unroll
            for (int j = 0; j < 4; j++) {
                int n = bq + c * 4 + j;
                Bg[n * KT + bk] = (__bf16)gv[j];
                Bu[n * KT + bk] = (__bf16)uv[j];
            }
        }
        cp_wait();
        __syncthreads();

#pragma unroll
        for (int ks = 0; ks < 2; ks++) {
            v16bf afr[2], gfr[2], ufr[2];
#pragma unroll
            for (int mt = 0; mt < 2; mt++)
                afr[mt] = make_afrag(As + (wm * 32 + mt * 16 + mrow) * KT + ks * 32, khalf);
#pragma unroll
            for (int nt = 0; nt < 2; nt++) {
                int nn = wn * 32 + nt * 16 + mrow;
                gfr[nt] = *(const v16bf*)(Bg + nn * KT + ks * 32 + khalf * 16);
                ufr[nt] = *(const v16bf*)(Bu + nn * KT + ks * 32 + khalf * 16);
            }
#pragma unroll
            for (int mt = 0; mt < 2; mt++)
#pragma unroll
                for (int nt = 0; nt < 2; nt++) {
                    accG[mt][nt] = __builtin_amdgcn_wmma_f32_16x16x32_bf16(
                        false, afr[mt], false, gfr[nt], (short)0, accG[mt][nt], false, false);
                    accU[mt][nt] = __builtin_amdgcn_wmma_f32_16x16x32_bf16(
                        false, afr[mt], false, ufr[nt], (short)0, accU[mt][nt], false, false);
                }
        }
    }

#pragma unroll
    for (int mt = 0; mt < 2; mt++)
#pragma unroll
        for (int nt = 0; nt < 2; nt++) {
            int n = n0 + wn * 32 + nt * 16 + mrow;
#pragma unroll
            for (int r = 0; r < 8; r++) {
                int m = wm * 32 + mt * 16 + khalf * 8 + r;
                int grow = row0 + m;
                if (grow < cnt) {
                    float s = accG[mt][nt][r];
                    float u = accU[mt][nt][r];
                    float hv = s * fast_rcp(1.f + __expf(-s)) * u;
                    hout[(size_t)grow * ID + n] = (__bf16)hv;
                }
            }
        }
}

// ----------------------------------------------------------- GEMM 2: down

__global__ __launch_bounds__(256)
void k_gemm_down(const __bf16* __restrict__ hA, const float* __restrict__ dw,
                 const int* __restrict__ list, const int* __restrict__ cnt_ptr,
                 const float* __restrict__ wts, const int* __restrict__ slots,
                 float* __restrict__ out, float* __restrict__ slotbuf) {
    int tid = threadIdx.x;
    int cnt = cnt_ptr ? *cnt_ptr : TOK;
    int row0 = blockIdx.y * 128;
    if (row0 >= cnt || cnt == 0) return;
    int n0 = blockIdx.x * 64;

    __shared__ __align__(32) __bf16 As[128 * KT];
    __shared__ __align__(32) __bf16 Bs[64 * KT];

    int lane = tid & 31;
    int wave = tid >> 5;
    int wm = wave & 3;
    int wn = wave >> 2;
    int mrow = lane & 15;
    int khalf = lane >> 4;

    v8f acc[2][2];
#pragma unroll
    for (int a = 0; a < 2; a++)
#pragma unroll
        for (int b = 0; b < 2; b++) {
            v8f z = {0.f, 0.f, 0.f, 0.f, 0.f, 0.f, 0.f, 0.f};
            acc[a][b] = z;
        }

    int ar = tid >> 1;
    int ah = (tid & 1) * 32;
    int aridx = row0 + ar; if (aridx >= cnt) aridx = cnt - 1;
    int bk = tid >> 2;
    int bq = (tid & 3) * 16;

    for (int k0 = 0; k0 < ID; k0 += KT) {
        const float* dp = dw + (size_t)(k0 + bk) * HD + n0 + bq;
        float4 d[4];
#pragma unroll
        for (int c = 0; c < 4; c++) d[c] = ((const float4*)dp)[c];
        if (k0 + KT < ID)
            __builtin_prefetch(dw + (size_t)(k0 + KT + bk) * HD + n0 + bq, 0, 1);
        __syncthreads();
        {
            const __bf16* asrc = hA + (size_t)aridx * ID + k0 + ah;
            __bf16* adst = As + ar * KT + ah;
#pragma unroll
            for (int c = 0; c < 4; c++)
                cp16(adst + c * 8, asrc + c * 8);
        }
#pragma unroll
        for (int c = 0; c < 4; c++) {
            float dv[4] = {d[c].x, d[c].y, d[c].z, d[c].w};
#pragma unroll
            for (int j = 0; j < 4; j++) {
                int n = bq + c * 4 + j;
                Bs[n * KT + bk] = (__bf16)dv[j];
            }
        }
        cp_wait();
        __syncthreads();

#pragma unroll
        for (int ks = 0; ks < 2; ks++) {
            v16bf afr[2], bfr[2];
#pragma unroll
            for (int mt = 0; mt < 2; mt++)
                afr[mt] = make_afrag(As + (wm * 32 + mt * 16 + mrow) * KT + ks * 32, khalf);
#pragma unroll
            for (int nt = 0; nt < 2; nt++) {
                int nn = wn * 32 + nt * 16 + mrow;
                bfr[nt] = *(const v16bf*)(Bs + nn * KT + ks * 32 + khalf * 16);
            }
#pragma unroll
            for (int mt = 0; mt < 2; mt++)
#pragma unroll
                for (int nt = 0; nt < 2; nt++)
                    acc[mt][nt] = __builtin_amdgcn_wmma_f32_16x16x32_bf16(
                        false, afr[mt], false, bfr[nt], (short)0, acc[mt][nt], false, false);
        }
    }

#pragma unroll
    for (int mt = 0; mt < 2; mt++)
#pragma unroll
        for (int nt = 0; nt < 2; nt++) {
            int n = n0 + wn * 32 + nt * 16 + mrow;
#pragma unroll
            for (int r = 0; r < 8; r++) {
                int m = wm * 32 + mt * 16 + khalf * 8 + r;
                int grow = row0 + m;
                if (grow < cnt) {
                    float v = acc[mt][nt][r];
                    if (!list) {
                        out[(size_t)grow * HD + n] = v;           // shared expert
                    } else {
                        int tok = list[grow];
                        float w = wts[grow];
                        int sl = slots[grow];
                        slotbuf[((size_t)sl * TOK + tok) * HD + n] = w * v;
                    }
                }
            }
        }
}

// ---------------------------------------------------------------- combine

__global__ __launch_bounds__(256)
void k_combine(float* __restrict__ out, const float* __restrict__ slotbuf) {
    int i = (blockIdx.x * 256 + threadIdx.x) * 4;
    float4 o = *(float4*)(out + i);
    float4 a = *(const float4*)(slotbuf + i);
    float4 b = *(const float4*)(slotbuf + (size_t)TOK * HD + i);
    o.x += a.x + b.x; o.y += a.y + b.y; o.z += a.z + b.z; o.w += a.w + b.w;
    *(float4*)(out + i) = o;
}

// ----------------------------------------------------------------- launch

extern "C" void kernel_launch(void* const* d_in, const int* in_sizes, int n_in,
                              void* d_out, int out_size, void* d_ws, size_t ws_size,
                              hipStream_t stream) {
    const float* x     = (const float*)d_in[0];
    const float* rw    = (const float*)d_in[1];
    const float* rbias = (const float*)d_in[2];
    const float* sg    = (const float*)d_in[3];
    const float* su    = (const float*)d_in[4];
    const float* sd    = (const float*)d_in[5];
    const float* rg    = (const float*)d_in[6];
    const float* ru    = (const float*)d_in[7];
    const float* rd    = (const float*)d_in[8];
    float* out = (float*)d_out;
    (void)in_sizes; (void)n_in; (void)out_size; (void)ws_size;

    char* ws = (char*)d_ws;
    size_t off = 0;
    auto take = [&](size_t bytes) {
        char* p = ws + off;
        off = (off + bytes + 255) & ~(size_t)255;
        return p;
    };
    __bf16* xb      = (__bf16*)take((size_t)TOK * HD * 2);
    __bf16* hbuf    = (__bf16*)take((size_t)TOK * ID * 2);
    float*  slotbuf = (float*) take((size_t)2 * TOK * HD * 4);
    int*    lists   = (int*)   take((size_t)NE * TOK * 4);
    float*  wts     = (float*) take((size_t)NE * TOK * 4);
    int*    slots   = (int*)   take((size_t)NE * TOK * 4);
    int*    counts  = (int*)   take(64);

    k_init_counts<<<1, 32, 0, stream>>>(counts);
    k_cvt_bf16<<<(TOK * HD) / 1024, 256, 0, stream>>>(x, xb, TOK * HD);
    k_router<<<TOK / 8, 256, 0, stream>>>(x, rw, rbias, counts, lists, wts, slots);

    dim3 g1(ID / 64, TOK / 128);   // (44, 32)
    dim3 g2(HD / 64, TOK / 128);   // (16, 32)

    // shared expert: identity token list, writes out with '='
    k_gemm_gateup<<<g1, 256, 0, stream>>>(xb, sg, su, nullptr, nullptr, hbuf);
    k_gemm_down<<<g2, 256, 0, stream>>>(hbuf, sd, nullptr, nullptr,
                                        nullptr, nullptr, out, nullptr);

    // routed experts, sequential on stream (hbuf reused)
    for (int e = 0; e < NE; e++) {
        const float* gwp = rg + (size_t)e * HD * ID;
        const float* uwp = ru + (size_t)e * HD * ID;
        const float* dwp = rd + (size_t)e * ID * HD;
        k_gemm_gateup<<<g1, 256, 0, stream>>>(xb, gwp, uwp,
                                              lists + e * TOK, counts + e, hbuf);
        k_gemm_down<<<g2, 256, 0, stream>>>(hbuf, dwp,
                                            lists + e * TOK, counts + e,
                                            wts + e * TOK, slots + e * TOK,
                                            nullptr, slotbuf);
    }

    k_combine<<<(TOK * HD) / 1024, 256, 0, stream>>>(out, slotbuf);
}
